// ELTM_28389733827266
// MI455X (gfx1250) — matrix operand order repaired and checked by reading, not compiled, not run on hardware
//
#include <hip/hip_runtime.h>
#include <math.h>

// ---------------------------------------------------------------------------
// ELTM forward for MI455X (gfx1250): bf16 WMMA (16x16x32, f32 accum),
// weights packed/fused so each of the 512 sequential steps is 3 dependent
// GEMM kernels per layer. Weights (~11MB bf16) stay L2-resident.
// ---------------------------------------------------------------------------

typedef __attribute__((ext_vector_type(16))) __bf16 v16bf;
typedef __attribute__((ext_vector_type(8)))  float  v8f;
typedef unsigned short u16b;

#define T_STEPS 512
#define BATCH   128
#define INSZ    256
#define HID     512

// ---- bf16 <-> f32 via bit ops (no dependence on __bf16 scalar arithmetic) --
__device__ __forceinline__ u16b f2bf(float f) {
  union { float f; unsigned u; } v; v.f = f;
  unsigned r = v.u + 0x7FFFu + ((v.u >> 16) & 1u);   // round-to-nearest-even
  return (u16b)(r >> 16);
}
__device__ __forceinline__ float bf2f(u16b h) {
  union { unsigned u; float f; } v; v.u = ((unsigned)h) << 16;
  return v.f;
}
__device__ __forceinline__ float sigmoidf_(float x) {
  return 1.0f / (1.0f + __expf(-x));
}

// ---- WMMA operand loaders (per CDNA5 ISA 7.12.2 VGPR layouts) --------------
// A (16x32 bf16, row-major src, lane = M row; K chunks {0..7,16..23} for
// lanes 0-15 and {8..15,24..31} for lanes 16-31).
__device__ __forceinline__ v16bf load_tile_A(const u16b* tile, int lda) {
  int lane = threadIdx.x & 31;
  int r    = lane & 15;
  int kh   = (lane >> 4) << 3;                 // 0 or 8
  const u16b* p = tile + (size_t)r * lda + kh;
  union { uint4 q[2]; v16bf v; } u;
  u.q[0] = *(const uint4*)(p);
  u.q[1] = *(const uint4*)(p + 16);
  return u.v;
}
// B (32x16 bf16). Weight stored [N][K] row-major (i.e. W as given, since the
// reference computes in @ W.T). Lane = N column; K chunks {0..7,16..23} for
// lanes 0-15 and {8..15,24..31} for lanes 16-31 (same as A).
__device__ __forceinline__ v16bf load_tile_B(const u16b* tile, int ldb) {
  int lane = threadIdx.x & 31;
  int n    = lane & 15;
  int kh   = (lane >> 4) << 3;                 // 0 or 8
  const u16b* p = tile + (size_t)n * ldb + kh;
  union { uint4 q[2]; v16bf v; } u;
  u.q[0] = *(const uint4*)(p);
  u.q[1] = *(const uint4*)(p + 16);
  return u.v;
}

__device__ __forceinline__ v8f wmma_bf16(v16bf a, v16bf b, v8f c) {
  // 8 args: (neg_a, A, neg_b, B, c_mod, C, reuse_a, reuse_b)
  return __builtin_amdgcn_wmma_f32_16x16x32_bf16(false, a, false, b,
                                                 (short)0, c, false, false);
}

// C/D layout: VGPR r holds row (m0 + r) for lanes 0-15, row (m0+8+r) for
// lanes 16-31; column = n0 + (lane & 15).

// ---------------------------------------------------------------------------
// Packing kernels (fp32 inputs -> packed bf16 weight layouts, run each call)
// ---------------------------------------------------------------------------
__global__ void k_cvt_bf16(u16b* __restrict__ dst, const float* __restrict__ src, int n) {
  int i = blockIdx.x * blockDim.x + threadIdx.x;
  if (i < n) dst[i] = f2bf(src[i]);
}

// Wbig [2048][K], K = 512 + Kx.  Rows: [0,1024) gates (k<512: W_hidden_in,
// k>=512: W_input_in); [1024,1536) xt (k>=512: W_only_x); [1536,2048) hh
// (k<512: W_only_h).  Zero elsewhere.
__global__ void k_pack_wbig(u16b* __restrict__ dst,
                            const float* __restrict__ Whi,
                            const float* __restrict__ Wii,
                            const float* __restrict__ Wox,
                            const float* __restrict__ Woh,
                            int Kx) {
  const int Kh = HID;
  const int K  = Kh + Kx;
  int i = blockIdx.x * blockDim.x + threadIdx.x;
  int total = 2048 * K;
  if (i >= total) return;
  int nrow = i / K, k = i - nrow * K;
  float v = 0.0f;
  if (nrow < 1024) {
    v = (k < Kh) ? Whi[(size_t)nrow * Kh + k] : Wii[(size_t)nrow * Kx + (k - Kh)];
  } else if (nrow < 1536) {
    if (k >= Kh) v = Wox[(size_t)(nrow - 1024) * Kx + (k - Kh)];
  } else {
    if (k < Kh)  v = Woh[(size_t)(nrow - 1536) * Kh + k];
  }
  dst[i] = f2bf(v);
}

// Wcat [512][1536]: k<512 W_input_out, <1024 W_hidden_out, else W_cand_out.
__global__ void k_pack_wcat(u16b* __restrict__ dst,
                            const float* __restrict__ Wio,
                            const float* __restrict__ Who,
                            const float* __restrict__ Wco) {
  int i = blockIdx.x * blockDim.x + threadIdx.x;
  int total = HID * 1536;
  if (i >= total) return;
  int nrow = i / 1536, k = i - nrow * 1536;
  float v;
  if (k < 512)       v = Wio[(size_t)nrow * HID + k];
  else if (k < 1024) v = Who[(size_t)nrow * HID + (k - 512)];
  else               v = Wco[(size_t)nrow * HID + (k - 1024)];
  dst[i] = f2bf(v);
}

// ---------------------------------------------------------------------------
// Per-time-step GEMM kernels. 256 waves per launch: wave w -> m-tile (w&7),
// n-tile (w>>3), M=128, base-N=512. blockDim=256 (8 waves), grid=32.
// ---------------------------------------------------------------------------

// stepA: [gates_i|gates_h|xt_pre|hh_pre] = [h(512)|x(Kx)] @ Wbig.T, then fused
// gate activations -> Astage[128][1536] = [xt | hh | ch] (bf16).
__global__ __launch_bounds__(256) void k_stepA(
    const u16b* __restrict__ hbuf,   // [128][512] bf16
    const u16b* __restrict__ xsrc,   // [128][Kx]  bf16
    int Kx,
    const u16b* __restrict__ Wbig,   // [2048][512+Kx] bf16
    const float* __restrict__ bias,  // [1024] fp32 (b_hidden_in)
    u16b* __restrict__ Astage)       // [128][1536] bf16
{
  const int Kh = HID;
  const int K  = Kh + Kx;
  int w    = (blockIdx.x * blockDim.x + threadIdx.x) >> 5;
  int lane = threadIdx.x & 31;
  int m0 = (w & 7) << 4;
  int n0 = (w >> 3) << 4;             // 0..496
  v8f a0 = {}, a1 = {}, a2 = {}, a3 = {};
  for (int k0 = 0; k0 < K; k0 += 32) {
    v16bf av;
    if (k0 < Kh) av = load_tile_A(hbuf + (size_t)m0 * Kh + k0, Kh);
    else         av = load_tile_A(xsrc + (size_t)m0 * Kx + (k0 - Kh), Kx);
    const u16b* Wb = Wbig + k0;
    v16bf b0 = load_tile_B(Wb + (size_t)(n0         ) * K, K);
    v16bf b1 = load_tile_B(Wb + (size_t)(n0 +  512  ) * K, K);
    v16bf b2 = load_tile_B(Wb + (size_t)(n0 + 1024  ) * K, K);
    v16bf b3 = load_tile_B(Wb + (size_t)(n0 + 1536  ) * K, K);
    a0 = wmma_bf16(av, b0, a0);
    a1 = wmma_bf16(av, b1, a1);
    a2 = wmma_bf16(av, b2, a2);
    a3 = wmma_bf16(av, b3, a3);
  }
  int n    = n0 + (lane & 15);
  int mrow = m0 + ((lane >> 4) << 3);
  float bi = bias[n], bh = bias[512 + n];
#pragma unroll
  for (int r = 0; r < 8; ++r) {
    int row = mrow + r;
    float gi = a0[r] + bi;
    float gh = a1[r] + bh;
    float ch = sigmoidf_(gi) * tanhf(gh);
    float xt = tanhf(a2[r]);
    float hh = tanhf(a3[r]);
    size_t base = (size_t)row * 1536 + n;
    Astage[base        ] = f2bf(xt);
    Astage[base +  512 ] = f2bf(hh);
    Astage[base + 1024 ] = f2bf(ch);
  }
}

// stepB: o = sigmoid([xt|hh|ch] @ Wcat.T + b_cand)  (K=1536, N=512)
__global__ __launch_bounds__(256) void k_stepB(
    const u16b* __restrict__ Astage, // [128][1536] bf16
    const u16b* __restrict__ Wcat,   // [512][1536] bf16
    const float* __restrict__ bcand, // [512] fp32
    u16b* __restrict__ obuf)         // [128][512] bf16
{
  int w    = (blockIdx.x * blockDim.x + threadIdx.x) >> 5;
  int lane = threadIdx.x & 31;
  int m0 = (w & 7) << 4;
  int n0 = (w >> 3) << 4;
  v8f acc = {};
  for (int k0 = 0; k0 < 1536; k0 += 32) {
    v16bf av = load_tile_A(Astage + (size_t)m0 * 1536 + k0, 1536);
    v16bf bv = load_tile_B(Wcat   + (size_t)n0 * 1536 + k0, 1536);
    acc = wmma_bf16(av, bv, acc);
  }
  int n    = n0 + (lane & 15);
  int mrow = m0 + ((lane >> 4) << 3);
  float b = bcand[n];
#pragma unroll
  for (int r = 0; r < 8; ++r) {
    int row = mrow + r;
    obuf[(size_t)row * HID + n] = f2bf(sigmoidf_(acc[r] + b));
  }
}

// stepC: alpha = o@Wa.T, beta = o@Wb.T (Wab = [Wa;Wb], 2 accumulators), then
// h_new = alpha*xt + beta*hh + (1-alpha-beta)*ch; write bf16 h (and fp32 out).
__global__ __launch_bounds__(256) void k_stepC(
    const u16b* __restrict__ obuf,   // [128][512] bf16
    const u16b* __restrict__ Wab,    // [1024][512] bf16
    const u16b* __restrict__ Astage, // [128][1536] bf16 (xt|hh|ch)
    u16b* __restrict__ hnew,         // [128][512] bf16
    float* __restrict__ outf)        // [128][512] fp32 or nullptr
{
  int w    = (blockIdx.x * blockDim.x + threadIdx.x) >> 5;
  int lane = threadIdx.x & 31;
  int m0 = (w & 7) << 4;
  int n0 = (w >> 3) << 4;
  v8f aA = {}, aB = {};
  for (int k0 = 0; k0 < HID; k0 += 32) {
    v16bf av = load_tile_A(obuf + (size_t)m0 * HID + k0, HID);
    v16bf b0 = load_tile_B(Wab + (size_t)(n0        ) * HID + k0, HID);
    v16bf b1 = load_tile_B(Wab + (size_t)(n0 + 512  ) * HID + k0, HID);
    aA = wmma_bf16(av, b0, aA);
    aB = wmma_bf16(av, b1, aB);
  }
  int n    = n0 + (lane & 15);
  int mrow = m0 + ((lane >> 4) << 3);
#pragma unroll
  for (int r = 0; r < 8; ++r) {
    int row = mrow + r;
    float alpha = aA[r], beta = aB[r];
    float gamma = 1.0f - alpha - beta;
    size_t base = (size_t)row * 1536 + n;
    float xt = bf2f(Astage[base        ]);
    float hh = bf2f(Astage[base +  512 ]);
    float ch = bf2f(Astage[base + 1024 ]);
    float h  = alpha * xt + beta * hh + gamma * ch;
    hnew[(size_t)row * HID + n] = f2bf(h);
    if (outf) outf[(size_t)row * HID + n] = h;
  }
}

// ---------------------------------------------------------------------------
// Final pass: SinReLU + LayerNorm per (t,b) row of d_out. One wave per row,
// 16 elems/lane, wave32 __shfl_xor reductions.
// ---------------------------------------------------------------------------
__global__ __launch_bounds__(256) void k_post(
    float* __restrict__ out, const float* __restrict__ gamma,
    const float* __restrict__ beta, int nrows)
{
  int w    = (blockIdx.x * blockDim.x + threadIdx.x) >> 5;
  int lane = threadIdx.x & 31;
  if (w >= nrows) return;
  float* p = out + (size_t)w * HID + lane * 16;
  float v[16];
  float s = 0.0f;
#pragma unroll
  for (int i = 0; i < 16; i += 4) {
    float4 q = *(const float4*)(p + i);
    v[i + 0] = q.x; v[i + 1] = q.y; v[i + 2] = q.z; v[i + 3] = q.w;
  }
#pragma unroll
  for (int i = 0; i < 16; ++i) {
    float x = v[i];
    float y = (x >= 0.0f) ? (x + __sinf(x)) : 0.0f;   // SINRELU_V = 1.0
    v[i] = y;
    s += y;
  }
#pragma unroll
  for (int off = 16; off > 0; off >>= 1) s += __shfl_xor(s, off, 32);
  float mu = s * (1.0f / HID);
  float ss = 0.0f;
#pragma unroll
  for (int i = 0; i < 16; ++i) { float d = v[i] - mu; ss += d * d; }
#pragma unroll
  for (int off = 16; off > 0; off >>= 1) ss += __shfl_xor(ss, off, 32);
  float rstd = rsqrtf(ss * (1.0f / HID) + 1e-5f);
#pragma unroll
  for (int i = 0; i < 16; ++i) {
    int col = lane * 16 + i;
    v[i] = (v[i] - mu) * rstd * gamma[col] + beta[col];
  }
#pragma unroll
  for (int i = 0; i < 16; i += 4) {
    float4 q; q.x = v[i]; q.y = v[i + 1]; q.z = v[i + 2]; q.w = v[i + 3];
    *(float4*)(p + i) = q;
  }
}

// ---------------------------------------------------------------------------
// Host launcher
// ---------------------------------------------------------------------------
struct CellPtrs {
  const float *Wii, *Whi, *bhi, *Wox, *Woh, *Wio, *Who, *Wco, *bco, *Wa, *Wb;
};

extern "C" void kernel_launch(void* const* d_in, const int* in_sizes, int n_in,
                              void* d_out, int out_size, void* d_ws, size_t ws_size,
                              hipStream_t stream) {
  (void)n_in; (void)out_size; (void)ws_size;
  auto F = [&](int i) { return (const float*)d_in[i]; };

  const float* x = nullptr;
  const float *ln_scale = nullptr, *ln_bias = nullptr;
  CellPtrs c[2];

  const int X_ELEMS = T_STEPS * BATCH * INSZ;  // 16777216
  if (in_sizes[0] == X_ELEMS) {
    // Python insertion order: x, cells[0]{insertion keys}, cells[1], ln_scale, ln_bias
    x = F(0);
    for (int l = 0; l < 2; ++l) {
      int b = 1 + l * 11;
      c[l].Wii = F(b + 0);  c[l].Whi = F(b + 1);  c[l].bhi = F(b + 2);
      c[l].Wox = F(b + 3);  c[l].Woh = F(b + 4);  c[l].Wio = F(b + 5);
      c[l].Who = F(b + 6);  c[l].Wco = F(b + 7);  c[l].bco = F(b + 8);
      c[l].Wa  = F(b + 9);  c[l].Wb  = F(b + 10);
    }
    ln_scale = F(23); ln_bias = F(24);
  } else {
    // jax tree-leaves (sorted keys): cells[0..1]{sorted keys}, ln_bias, ln_scale, x
    for (int l = 0; l < 2; ++l) {
      int b = l * 11;
      c[l].Wa  = F(b + 0);  c[l].Wb  = F(b + 1);  c[l].Wco = F(b + 2);
      c[l].Whi = F(b + 3);  c[l].Who = F(b + 4);  c[l].Wii = F(b + 5);
      c[l].Wio = F(b + 6);  c[l].Woh = F(b + 7);  c[l].Wox = F(b + 8);
      c[l].bco = F(b + 9);  c[l].bhi = F(b + 10);
    }
    ln_bias = F(22); ln_scale = F(23); x = F(24);
  }

  // ---- carve workspace (all offsets 256B aligned) ----
  size_t off = 0;
  auto carve = [&](size_t elems) {
    void* p = (char*)d_ws + off;
    off += ((elems * sizeof(u16b)) + 255) & ~(size_t)255;
    return (u16b*)p;
  };
  u16b* xbf    = carve((size_t)X_ELEMS);           // x as bf16
  u16b* Wbig0  = carve((size_t)2048 * 768);
  u16b* Wcat0  = carve((size_t)HID * 1536);
  u16b* Wab0   = carve((size_t)1024 * HID);
  u16b* Wbig1  = carve((size_t)2048 * 1024);
  u16b* Wcat1  = carve((size_t)HID * 1536);
  u16b* Wab1   = carve((size_t)1024 * HID);
  u16b* h0buf  = carve((size_t)BATCH * HID);
  u16b* h1buf  = carve((size_t)BATCH * HID);
  u16b* Ast0   = carve((size_t)BATCH * 1536);
  u16b* Ast1   = carve((size_t)BATCH * 1536);
  u16b* o0     = carve((size_t)BATCH * HID);
  u16b* o1     = carve((size_t)BATCH * HID);

  auto blocks = [](int n) { return (n + 255) / 256; };

  // ---- phase 0: convert / pack (parallel, weights then stay L2-resident) ---
  k_cvt_bf16<<<blocks(X_ELEMS), 256, 0, stream>>>(xbf, x, X_ELEMS);
  k_pack_wbig<<<blocks(2048 * 768), 256, 0, stream>>>(Wbig0, c[0].Whi, c[0].Wii,
                                                      c[0].Wox, c[0].Woh, INSZ);
  k_pack_wcat<<<blocks(HID * 1536), 256, 0, stream>>>(Wcat0, c[0].Wio, c[0].Who, c[0].Wco);
  k_cvt_bf16<<<blocks(HID * HID), 256, 0, stream>>>(Wab0, c[0].Wa, HID * HID);
  k_cvt_bf16<<<blocks(HID * HID), 256, 0, stream>>>(Wab0 + (size_t)HID * HID, c[0].Wb, HID * HID);
  k_pack_wbig<<<blocks(2048 * 1024), 256, 0, stream>>>(Wbig1, c[1].Whi, c[1].Wii,
                                                       c[1].Wox, c[1].Woh, HID);
  k_pack_wcat<<<blocks(HID * 1536), 256, 0, stream>>>(Wcat1, c[1].Wio, c[1].Who, c[1].Wco);
  k_cvt_bf16<<<blocks(HID * HID), 256, 0, stream>>>(Wab1, c[1].Wa, HID * HID);
  k_cvt_bf16<<<blocks(HID * HID), 256, 0, stream>>>(Wab1 + (size_t)HID * HID, c[1].Wb, HID * HID);

  hipMemsetAsync(h0buf, 0, (size_t)BATCH * HID * sizeof(u16b), stream);
  hipMemsetAsync(h1buf, 0, (size_t)BATCH * HID * sizeof(u16b), stream);

  // ---- phase 1: sequential scan over T (the latency-bound critical path) ---
  float* outf = (float*)d_out;
  for (int t = 0; t < T_STEPS; ++t) {
    const u16b* xt0 = xbf + (size_t)t * BATCH * INSZ;
    // layer 0
    k_stepA<<<32, 256, 0, stream>>>(h0buf, xt0, INSZ, Wbig0, c[0].bhi, Ast0);
    k_stepB<<<32, 256, 0, stream>>>(Ast0, Wcat0, c[0].bco, o0);
    k_stepC<<<32, 256, 0, stream>>>(o0, Wab0, Ast0, h0buf, nullptr);
    // layer 1 (input = freshly updated h0)
    k_stepA<<<32, 256, 0, stream>>>(h1buf, h0buf, HID, Wbig1, c[1].bhi, Ast1);
    k_stepB<<<32, 256, 0, stream>>>(Ast1, Wcat1, c[1].bco, o1);
    k_stepC<<<32, 256, 0, stream>>>(o1, Wab1, Ast1, h1buf,
                                    outf + (size_t)t * BATCH * HID);
  }

  // ---- phase 2: SinReLU + LayerNorm over all rows of d_out ----
  int nrows = T_STEPS * BATCH;                   // 65536 rows, 1 wave each
  k_post<<<(nrows * 32 + 255) / 256, 256, 0, stream>>>(outf, ln_scale, ln_bias, nrows);
}
